// ALSTM_5196910428287
// MI455X (gfx1250) — compile-verified, matrix operand order checked by reading, not run on hardware
//
#include <hip/hip_runtime.h>

typedef __bf16 bf16;
typedef __attribute__((ext_vector_type(16))) __bf16 v16bf;
typedef __attribute__((ext_vector_type(8)))  __bf16 v8bf;
typedef __attribute__((ext_vector_type(8)))  float  v8f;

#define T_STEPS   512
#define BATCH     32
#define IN_SZ     128
#define HID       512
#define OUT_SZ    128
#define MAXP      8
#define ACT_EPS   0.01f

// fragment region sizes (bf16 elements); each 16(N) x 32(K) fragment = 512 bf16
#define NIH_FR    (4  * 128 * 512)   // W_ih: K=128 (4 kblks), N=2048 (128 ntiles)
#define NHH_FR    (16 * 128 * 512)   // W_hh: K=512, N=2048
#define NOUT_FR   (16 * 8   * 512)   // W_out: K=512, N=128 (8 ntiles)

__device__ __forceinline__ v8f wmma_bf16(v16bf a, v16bf b, v8f c) {
    return __builtin_amdgcn_wmma_f32_16x16x32_bf16(false, a, false, b, (short)0, c,
                                                   false, false);
}

__device__ __forceinline__ float sigf(float x) {
    return 1.0f / (1.0f + __expf(-x));
}

// A-matrix (16x32 bf16) fragment from row-major LDS buffer, per ISA layout:
// lane L: M = m0 + (L&15); h=L>>4; elems 0..7 -> K=k0+8h+i, elems 8..15 -> K=k0+16+8h+i
__device__ __forceinline__ v16bf load_a(const bf16* base, int ldk, int m0, int k0, int lane) {
    int h   = lane >> 4;
    int row = m0 + (lane & 15);
    const v8bf lo = *(const v8bf*)(base + row * ldk + k0 + h * 8);
    const v8bf hi = *(const v8bf*)(base + row * ldk + k0 + 16 + h * 8);
    v16bf a;
#pragma unroll
    for (int i = 0; i < 8; ++i) { a[i] = lo[i]; a[8 + i] = hi[i]; }
    return a;
}

// ---------------- weight prep: f32 -> bf16 pre-swizzled B fragments ----------------
__global__ void prep_frags(const float* __restrict__ Wih,
                           const float* __restrict__ Whh,
                           const float* __restrict__ Wout,
                           bf16* __restrict__ fih,
                           bf16* __restrict__ fhh,
                           bf16* __restrict__ fout) {
    int idx = blockIdx.x * blockDim.x + threadIdx.x;
    const float* W; bf16* dst; int ntiles, ldw, coff, lidx;
    if (idx < NIH_FR)                     { W = Wih;  dst = fih;  ntiles = 128; ldw = 129; coff = 1; lidx = idx; }
    else if (idx < NIH_FR + NHH_FR)       { W = Whh;  dst = fhh;  ntiles = 128; ldw = 512; coff = 0; lidx = idx - NIH_FR; }
    else if (idx < NIH_FR + NHH_FR + NOUT_FR)
                                          { W = Wout; dst = fout; ntiles = 8;   ldw = 512; coff = 0; lidx = idx - NIH_FR - NHH_FR; }
    else return;
    int frag = lidx >> 9, s = lidx & 511;
    int L = s >> 4, e = s & 15;
    int kb = frag / ntiles, nt = frag % ntiles;
    int n = nt * 16 + (L & 15);
    int h = L >> 4;
    int k = kb * 32 + ((e < 8) ? (h * 8 + e) : (16 + h * 8 + (e - 8)));
    dst[lidx] = (bf16)W[n * ldw + k + coff];
}

// ---------------- persistent recurrent ACT-LSTM kernel (1 workgroup, 32 waves) ------
__global__ __launch_bounds__(1024) void act_lstm(
    const float* __restrict__ x,     const float* __restrict__ h0,
    const float* __restrict__ c0,    const float* __restrict__ Wih,
    const float* __restrict__ bih,   const float* __restrict__ bhh,
    const float* __restrict__ bout,  const float* __restrict__ Whalt,
    const float* __restrict__ bhalt, const bf16* __restrict__ fih,
    const bf16* __restrict__ fhh,    const bf16* __restrict__ fout,
    float* __restrict__ outs,        float* __restrict__ ponds) {

    __shared__ bf16  h_bf[BATCH * HID];        // 32 KB  masked hidden state (bf16)
    __shared__ bf16  x_bf[BATCH * IN_SZ];      // 8 KB   current x_t (bf16)
    __shared__ float whalt_s[HID];             // 2 KB
    __shared__ float acc_s[BATCH * HID];       // 64 KB  halt-weighted hidden accum
    __shared__ float acc_o[BATCH * OUT_SZ];    // 16 KB  halt-weighted output accum
    __shared__ float out_tmp[BATCH * OUT_SZ];  // 16 KB  per-step output projection
    __shared__ float sc_halted[32], sc_halt[32], sc_cum[32], sc_pond[32];
    __shared__ float sc_p[32], sc_peff[32], sc_live[32];
    __shared__ int   sDone;

    const int tid  = threadIdx.x;
    const int wid  = tid >> 5;
    const int lane = tid & 31;
    const int half = lane >> 4;
    const int jcol = (wid << 4) + (lane & 15);   // hidden column owned by this lane

    const v16bf* fihv  = (const v16bf*)fih;
    const v16bf* fhhv  = (const v16bf*)fhh;
    const v16bf* foutv = (const v16bf*)fout;

    // per-lane constants
    float gbase[4], gflag[4];
#pragma unroll
    for (int g = 0; g < 4; ++g) {
        int n = g * HID + jcol;
        gbase[g] = bih[n] + bhh[n];
        gflag[g] = Wih[n * 129 + 0];       // "flag" column (first ponder step only)
    }
    float creg[2][8];                       // persistent cell state c[:, jcol]
#pragma unroll
    for (int m = 0; m < 2; ++m)
#pragma unroll
        for (int r = 0; r < 8; ++r)
            creg[m][r] = c0[(16 * m + 8 * half + r) * HID + jcol];

    float boutreg = 0.0f, bhreg = 0.0f;
    if (wid < 16) boutreg = bout[(wid & 7) * 16 + (lane & 15)];
    if (wid == 16) bhreg = bhalt[0];

    for (int e = tid; e < BATCH * HID; e += 1024) h_bf[e] = (bf16)h0[e];
    if (tid < HID) whalt_s[tid] = Whalt[tid];
    __syncthreads();

    for (int t = 0; t < T_STEPS; ++t) {
        // ---- timestep init ----
        for (int e = tid; e < BATCH * IN_SZ; e += 1024) {
            x_bf[e]  = (bf16)x[(size_t)t * BATCH * IN_SZ + e];
            acc_o[e] = 0.0f;
        }
        for (int e = tid; e < BATCH * HID; e += 1024) acc_s[e] = 0.0f;
        if (wid == 31) {
            sc_halted[lane] = 0.0f; sc_halt[lane] = 0.0f; sc_cum[lane] = 0.0f;
            sc_pond[lane]   = 0.0f; sc_live[lane] = 1.0f;
            if (lane == 0) sDone = 0;
        }
        __syncthreads();

        // ---- ponder loop ----
        for (int n = 0; n < MAXP; ++n) {
            __syncthreads();
            if (sDone) break;

            // phase A: gate pre-activations G = [flag,x]@Wih^T + h@Whh^T
            v8f acc[4][2];
#pragma unroll
            for (int g = 0; g < 4; ++g) { acc[g][0] = (v8f){}; acc[g][1] = (v8f){}; }

#pragma unroll
            for (int kb = 0; kb < 4; ++kb) {            // x part, K = 128
                v16bf a0 = load_a(x_bf, IN_SZ, 0,  kb * 32, lane);
                v16bf a1 = load_a(x_bf, IN_SZ, 16, kb * 32, lane);
#pragma unroll
                for (int g = 0; g < 4; ++g) {
                    v16bf bm = fihv[(kb * 128 + (g * 32 + wid)) * 32 + lane];
                    acc[g][0] = wmma_bf16(a0, bm, acc[g][0]);
                    acc[g][1] = wmma_bf16(a1, bm, acc[g][1]);
                }
            }
#pragma unroll 4
            for (int kb = 0; kb < 16; ++kb) {           // h part, K = 512
                v16bf a0 = load_a(h_bf, HID, 0,  kb * 32, lane);
                v16bf a1 = load_a(h_bf, HID, 16, kb * 32, lane);
#pragma unroll
                for (int g = 0; g < 4; ++g) {
                    v16bf bm = fhhv[(kb * 128 + (g * 32 + wid)) * 32 + lane];
                    acc[g][0] = wmma_bf16(a0, bm, acc[g][0]);
                    acc[g][1] = wmma_bf16(a1, bm, acc[g][1]);
                }
            }

            // cell update in registers; gate order i,f,g,o ; write masked h after barrier
            float hm[2][8];
            {
                float gb0 = gbase[0] + (n == 0 ? gflag[0] : 0.0f);
                float gb1 = gbase[1] + (n == 0 ? gflag[1] : 0.0f);
                float gb2 = gbase[2] + (n == 0 ? gflag[2] : 0.0f);
                float gb3 = gbase[3] + (n == 0 ? gflag[3] : 0.0f);
#pragma unroll
                for (int m = 0; m < 2; ++m)
#pragma unroll
                    for (int r = 0; r < 8; ++r) {
                        int row = 16 * m + 8 * half + r;
                        float iv = acc[0][m][r] + gb0;
                        float fv = acc[1][m][r] + gb1;
                        float gv = acc[2][m][r] + gb2;
                        float ov = acc[3][m][r] + gb3;
                        float cn = sigf(fv) * creg[m][r] + sigf(iv) * tanhf(gv);
                        creg[m][r] = cn;
                        hm[m][r]   = sigf(ov) * tanhf(cn) * sc_live[row];
                    }
            }

            // phase B: publish masked h (hm registers die here)
            __syncthreads();
#pragma unroll
            for (int m = 0; m < 2; ++m)
#pragma unroll
                for (int r = 0; r < 8; ++r)
                    h_bf[(16 * m + 8 * half + r) * HID + jcol] = (bf16)hm[m][r];
            __syncthreads();

            // phase C: output projection (waves 0..15) + halting prob (wave 16)
            if (wid < 16) {
                int mt = wid >> 3, nto = wid & 7;
                v8f oa = (v8f){};
#pragma unroll 4
                for (int kb = 0; kb < 16; ++kb) {
                    v16bf a  = load_a(h_bf, HID, mt * 16, kb * 32, lane);
                    v16bf bm = foutv[(kb * 8 + nto) * 32 + lane];
                    oa = wmma_bf16(a, bm, oa);
                }
                int col = nto * 16 + (lane & 15);
#pragma unroll
                for (int r = 0; r < 8; ++r)
                    out_tmp[(16 * mt + 8 * half + r) * OUT_SZ + col] = oa[r] + boutreg;
            } else if (wid == 16) {
                float s = bhreg;
                const bf16* hrow = &h_bf[lane * HID];
                for (int j0 = 0; j0 < HID; j0 += 8) {
                    v8bf hv = *(const v8bf*)(hrow + j0);
#pragma unroll
                    for (int i = 0; i < 8; ++i) s += (float)hv[i] * whalt_s[j0 + i];
                }
                sc_p[lane] = sigf(s);
            }
            __syncthreads();

            // phase D: ACT bookkeeping (one wave, lane==batch)
            if (wid == 0) {
                float p      = sc_p[lane];
                float halted = sc_halted[lane];
                float halt   = sc_halt[lane];
                float cum    = sc_cum[lane];
                float pond   = sc_pond[lane];
                float live   = 1.0f - halted;
                float tmp    = (live * (halt + p) >= 1.0f - ACT_EPS) ? 1.0f : 0.0f;
                float rr     = tmp * (1.0f - cum);
                pond        += (rr + (float)(n + 1)) * tmp;
                float pn     = p * (1.0f - tmp) + rr;
                halt        += pn;
                pn          *= live;
                halted      += tmp;
                cum         += pn;
                sc_halted[lane] = halted; sc_halt[lane] = halt;
                sc_cum[lane]    = cum;    sc_pond[lane] = pond;
                sc_peff[lane]   = pn;     sc_live[lane] = 1.0f - halted;
                int alldone = __all(halted > 0.5f);
                if (lane == 0) sDone = alldone;
            }
            __syncthreads();

            // phase E: halt-prob weighted accumulation (flat cooperative, LDS)
            for (int e = tid; e < BATCH * OUT_SZ; e += 1024)
                acc_o[e] += out_tmp[e] * sc_peff[e >> 7];
            for (int e = tid; e < BATCH * HID; e += 1024)
                acc_s[e] += (float)h_bf[e] * sc_peff[e >> 9];
        } // ponder

        // ---- end of timestep: h <- acc_s, emit outputs ----
        __syncthreads();
        for (int e = tid; e < BATCH * HID; e += 1024) h_bf[e] = (bf16)acc_s[e];
        for (int e = tid; e < BATCH * OUT_SZ; e += 1024)
            outs[(size_t)t * BATCH * OUT_SZ + e] = acc_o[e];
        if (wid == 16) ponds[t * BATCH + lane] = sc_pond[lane];
        __syncthreads();
    } // t
}

extern "C" void kernel_launch(void* const* d_in, const int* in_sizes, int n_in,
                              void* d_out, int out_size, void* d_ws, size_t ws_size,
                              hipStream_t stream) {
    const float* x     = (const float*)d_in[0];
    const float* h0    = (const float*)d_in[1];
    const float* c0    = (const float*)d_in[2];
    const float* Wih   = (const float*)d_in[3];
    const float* Whh   = (const float*)d_in[4];
    const float* bih   = (const float*)d_in[5];
    const float* bhh   = (const float*)d_in[6];
    const float* Wout  = (const float*)d_in[7];
    const float* bout  = (const float*)d_in[8];
    const float* Whalt = (const float*)d_in[9];
    const float* bhalt = (const float*)d_in[10];

    bf16* fih  = (bf16*)d_ws;
    bf16* fhh  = fih + NIH_FR;
    bf16* fout = fhh + NHH_FR;

    int total = NIH_FR + NHH_FR + NOUT_FR;
    prep_frags<<<(total + 255) / 256, 256, 0, stream>>>(Wih, Whh, Wout, fih, fhh, fout);

    float* outs  = (float*)d_out;
    float* ponds = outs + (size_t)T_STEPS * BATCH * OUT_SZ;
    act_lstm<<<1, 1024, 0, stream>>>(x, h0, c0, Wih, bih, bhh, bout, Whalt, bhalt,
                                     fih, fhh, fout, outs, ponds);
}
